// VisionMemory_83872121357090
// MI455X (gfx1250) — compile-verified, matrix operand order checked by reading, not run on hardware
//
#include <hip/hip_runtime.h>
#include <hip/hip_bf16.h>

#define B_ 32
#define S_ 4096
#define D_ 256
#define M_ 512
#define ROWS_TOTAL (B_ * S_)   // 131072
#define TILE_ROWS 64           // rows per workgroup (4 waves x 16)
#define XLD 264                // x_bf LDS row stride (halves), padded
#define SLD 516                // scores LDS row stride (floats), padded
#define PLD 520                // p_bf LDS row stride (halves), padded
#define RBF_OFF 260            // float offset inside a score row where bf16 R lives
#define RBLD (SLD * 2)         // halves between consecutive rows for bf16 R view

#define X_BYTES (TILE_ROWS * XLD * 2)              // 33792
#define S_BYTES (TILE_ROWS * SLD * 4)              // 132096
#define P_BYTES (TILE_ROWS * PLD * 2)              // 66560
#define SMEM_BYTES (X_BYTES + S_BYTES + P_BYTES)   // 232448 < 320KB

typedef __attribute__((ext_vector_type(16))) __bf16 v16bf;
typedef __attribute__((ext_vector_type(8)))  __bf16 v8bf;
typedef __attribute__((ext_vector_type(4)))  __bf16 v4bf;
typedef __attribute__((ext_vector_type(8)))  float  v8f;
typedef __attribute__((ext_vector_type(4)))  float  v4f;

union FragU { v16bf v; v8bf h[2]; };

// Load one 16x32 bf16 WMMA operand fragment from a row-major (in K) source.
// ISA layout: lanes 0-15 -> halves 0..7 = K=k..k+7, halves 8..15 = K=k+16..k+23
//             lanes 16-31 -> same with K offset +8. 16-byte aligned b128 loads.
__device__ __forceinline__ v16bf load_frag(const __bf16* base, int ld, int k_base, int lane) {
  const int r = lane & 15, hi = lane >> 4;
  const __bf16* p = base + (size_t)r * ld + k_base + hi * 8;
  FragU f;
  f.h[0] = *(const v8bf*)(p);
  f.h[1] = *(const v8bf*)(p + 16);
  return f.v;
}

__device__ __forceinline__ v8f wmma_bf16(v16bf a, v16bf b, v8f c) {
  return __builtin_amdgcn_wmma_f32_16x16x32_bf16(false, a, false, b, (short)0, c, false, false);
}

// Convert L2-resident operands to bf16: memory, memory^T, Wg. 131072 threads.
__global__ __launch_bounds__(256)
void vm_prep(const float* __restrict__ mem, const float* __restrict__ wg,
             __bf16* __restrict__ mem_bf, __bf16* __restrict__ memT_bf,
             __bf16* __restrict__ wg_bf) {
  int i = blockIdx.x * 256 + threadIdx.x;       // 0 .. M_*D_-1 (also covers Wg)
  __bf16 b = (__bf16)mem[i];
  mem_bf[i] = b;
  int m = i >> 8, d = i & 255;                  // memory is 512x256
  memT_bf[d * M_ + m] = b;                      // memT is 256x512
  wg_bf[i] = (__bf16)wg[i];                     // Wg is 256x512
}

__global__ __launch_bounds__(128)
void vm_main(const float* __restrict__ x,
             const float* __restrict__ bg,
             const __bf16* __restrict__ mem_bf,
             const __bf16* __restrict__ memT_bf,
             const __bf16* __restrict__ wg_bf,
             float* __restrict__ out_enh,
             float* __restrict__ out_w) {
  extern __shared__ __align__(16) char smem[];
  __bf16* x_bf   = (__bf16*)smem;
  float*  scores = (float*)(smem + X_BYTES);
  __bf16* p_bf   = (__bf16*)(smem + X_BYTES + S_BYTES);

  const int tid   = threadIdx.x;
  const int lane  = tid & 31;
  const int w     = tid >> 5;
  const int lrow0 = w * 16;                       // this wave's 16-row tile
  const size_t rowbase = (size_t)blockIdx.x * TILE_ROWS;

  // ---- stage x tile (64x256 f32 -> bf16 LDS, padded stride) ----
  for (int i = tid * 4; i < TILE_ROWS * D_; i += 128 * 4) {
    int r = i >> 8, c = i & 255;
    v4f xv = *(const v4f*)(x + (rowbase + r) * D_ + c);
    __bf16* dst = x_bf + r * XLD + c;
    dst[0] = (__bf16)xv.x; dst[1] = (__bf16)xv.y;
    dst[2] = (__bf16)xv.z; dst[3] = (__bf16)xv.w;
  }
  __syncthreads();   // only workgroup-wide sync needed; rest is wave-private

  const __bf16* aX = x_bf + lrow0 * XLD;
  const int mo = 8 * (lane >> 4);
  const int nn = lane & 15;

  // ---- GEMM1: scores (16 x 512 per wave), K = 256 ----
  // Hoist all 8 A fragments (x, K=256) into registers: loaded once, used 8x.
  {
    v16bf afrag[8];
    #pragma unroll
    for (int kk = 0; kk < 8; ++kk) afrag[kk] = load_frag(aX, XLD, kk * 32, lane);

    for (int nc = 0; nc < 8; ++nc) {
      v8f acc[4] = {};
      #pragma unroll
      for (int kk = 0; kk < 8; ++kk) {
        #pragma unroll
        for (int t = 0; t < 4; ++t) {
          v16bf b = load_frag(mem_bf + (size_t)(nc * 64 + t * 16) * D_, D_, kk * 32, lane);
          acc[t] = wmma_bf16(afrag[kk], b, acc[t]);
        }
      }
      #pragma unroll
      for (int t = 0; t < 4; ++t)
        #pragma unroll
        for (int i = 0; i < 8; ++i)
          scores[(lrow0 + mo + i) * SLD + nc * 64 + t * 16 + nn] = acc[t][i];
    }
  }

  // ---- softmax over 512 cols; 2 lanes per row (halves), shfl_xor(16) combine ----
  // Fully vectorized: b128 LDS ops, b128 global weight stores, packed bf16 P stores.
  {
    const int r = lane & 15, half = lane >> 4;
    float* srow = scores + (lrow0 + r) * SLD + half * 256;
    float mx = -3.402823466e38f;
    #pragma unroll 8
    for (int c = 0; c < 256; c += 4) {
      v4f v = *(const v4f*)(srow + c);
      mx = fmaxf(mx, fmaxf(fmaxf(v.x, v.y), fmaxf(v.z, v.w)));
    }
    mx = fmaxf(mx, __shfl_xor(mx, 16, 32));
    float sum = 0.f;
    #pragma unroll 8
    for (int c = 0; c < 256; c += 4) {
      v4f v = *(const v4f*)(srow + c);
      v.x = __expf(v.x - mx); v.y = __expf(v.y - mx);
      v.z = __expf(v.z - mx); v.w = __expf(v.w - mx);
      *(v4f*)(srow + c) = v;
      sum += (v.x + v.y) + (v.z + v.w);
    }
    sum += __shfl_xor(sum, 16, 32);
    float inv = 1.f / sum;
    float* wrow = out_w + (rowbase + lrow0 + r) * (size_t)M_ + half * 256;
    __bf16* prow = p_bf + (lrow0 + r) * PLD + half * 256;
    #pragma unroll 8
    for (int c = 0; c < 256; c += 4) {
      v4f v = *(const v4f*)(srow + c);
      v4f wv; wv.x = v.x * inv; wv.y = v.y * inv; wv.z = v.z * inv; wv.w = v.w * inv;
      *(v4f*)(wrow + c) = wv;                    // weights output (f32)
      v4bf pb;
      pb.x = (__bf16)wv.x; pb.y = (__bf16)wv.y;
      pb.z = (__bf16)wv.z; pb.w = (__bf16)wv.w;
      *(v4bf*)(prow + c) = pb;                   // bf16 P for GEMM2
    }
  }

  // ---- GEMM2: R = P @ memory (16 x 256 per wave), K = 512 ----
  // Hoist all 16 P fragments; R overlays this wave's own score rows:
  // f32 at cols [0,256), bf16 at float-offset 260.
  {
    const __bf16* aP = p_bf + lrow0 * PLD;
    v16bf pfrag[16];
    #pragma unroll
    for (int kk = 0; kk < 16; ++kk) pfrag[kk] = load_frag(aP, PLD, kk * 32, lane);

    for (int nc = 0; nc < 4; ++nc) {
      v8f acc[4] = {};
      #pragma unroll
      for (int kk = 0; kk < 16; ++kk) {
        #pragma unroll
        for (int t = 0; t < 4; ++t) {
          v16bf b = load_frag(memT_bf + (size_t)(nc * 64 + t * 16) * M_, M_, kk * 32, lane);
          acc[t] = wmma_bf16(pfrag[kk], b, acc[t]);
        }
      }
      #pragma unroll
      for (int t = 0; t < 4; ++t)
        #pragma unroll
        for (int i = 0; i < 8; ++i) {
          int lrow = lrow0 + mo + i;
          int col  = nc * 64 + t * 16 + nn;
          float v  = acc[t][i];
          scores[lrow * SLD + col] = v;                                // R f32
          ((__bf16*)(scores + lrow * SLD + RBF_OFF))[col] = (__bf16)v; // R bf16
        }
    }
  }

  // ---- GEMM3: G = X@W1^T + R@W2^T ; epilogue: out = x + sigmoid(G+bg)*R ----
  {
    const __bf16* aR = (const __bf16*)(scores + RBF_OFF) + lrow0 * RBLD;
    v16bf axf[8], arf[8];
    #pragma unroll
    for (int kk = 0; kk < 8; ++kk) {
      axf[kk] = load_frag(aX, XLD, kk * 32, lane);
      arf[kk] = load_frag(aR, RBLD, kk * 32, lane);
    }

    for (int nc = 0; nc < 4; ++nc) {
      v8f acc[4] = {};
      #pragma unroll
      for (int kk = 0; kk < 8; ++kk) {
        #pragma unroll
        for (int t = 0; t < 4; ++t) {
          const __bf16* wrow = wg_bf + (size_t)(nc * 64 + t * 16) * (2 * D_);
          v16bf b1 = load_frag(wrow,      2 * D_, kk * 32, lane);   // W1[e][k]
          v16bf b2 = load_frag(wrow + D_, 2 * D_, kk * 32, lane);   // W2[e][k]
          acc[t] = wmma_bf16(axf[kk], b1, acc[t]);
          acc[t] = wmma_bf16(arf[kk], b2, acc[t]);
        }
      }
      #pragma unroll
      for (int i = 0; i < 8; ++i) {
        int lrow = lrow0 + mo + i;
        size_t grow = rowbase + (size_t)lrow;
        #pragma unroll
        for (int t = 0; t < 4; ++t) {
          int col = nc * 64 + t * 16 + nn;
          float tsum = acc[t][i] + bg[col];
          float g = 1.f / (1.f + __expf(-tsum));
          float rv = scores[lrow * SLD + col];       // R f32
          float xv = x[grow * D_ + col];
          out_enh[grow * D_ + col] = xv + g * rv;
        }
      }
    }
  }
}

extern "C" void kernel_launch(void* const* d_in, const int* in_sizes, int n_in,
                              void* d_out, int out_size, void* d_ws, size_t ws_size,
                              hipStream_t stream) {
  const float* x   = (const float*)d_in[0];
  const float* mem = (const float*)d_in[1];
  const float* Wg  = (const float*)d_in[2];
  const float* bg  = (const float*)d_in[3];

  __bf16* mem_bf  = (__bf16*)d_ws;                 // 512x256
  __bf16* memT_bf = mem_bf + (size_t)M_ * D_;      // 256x512
  __bf16* wg_bf   = memT_bf + (size_t)D_ * M_;     // 256x512

  float* out_enh = (float*)d_out;                          // B*S*D
  float* out_w   = out_enh + (size_t)ROWS_TOTAL * D_;      // B*S*M

  vm_prep<<<(M_ * D_) / 256, 256, 0, stream>>>(mem, Wg, mem_bf, memT_bf, wg_bf);
  vm_main<<<ROWS_TOTAL / TILE_ROWS, 128, SMEM_BYTES, stream>>>(
      x, bg, mem_bf, memT_bf, wg_bf, out_enh, out_w);
}
